// GroupConvolutionLayer2d_31404800869145
// MI455X (gfx1250) — matrix operand order repaired and checked by reading, not compiled, not run on hardware
//
#include <hip/hip_runtime.h>
#include <hip/hip_bf16.h>
#include <math.h>

// ---------------------------------------------------------------------------
// y[b, i, j] = inv_b * ( sum_n x[b,n]*lm[p,n]  -  mu_b * S_p ),  p = i*32+j
// S_p = sum_n lm_raw[p,n]  (Gaussian taps sum to 1; circular conv keeps sums)
// GEMM: M=16384 (b), N=1024 (p), K=1024 (n), f16 in, f32 accumulate (WMMA).
// ---------------------------------------------------------------------------

typedef _Float16 v16h __attribute__((ext_vector_type(16)));
typedef _Float16 v8h  __attribute__((ext_vector_type(8)));
typedef float    v8f  __attribute__((ext_vector_type(8)));

#define N_IN   1024
#define N_P    1024
#define M_ROWS 16384
#define FILT   33
#define SIGMA0 0.1f
#define EPS    1e-7f

#define BM 128
#define BN 64
#define BK 32
#define SPAD 40   // padded LDS row stride in halves (conflict-free b128 reads)

#if defined(__has_builtin)
#if __has_builtin(__builtin_amdgcn_global_load_async_to_lds_b128) && \
    __has_builtin(__builtin_amdgcn_s_wait_asynccnt)
#define HAVE_ASYNC_LDS 1
#endif
#endif

#ifdef HAVE_ASYNC_LDS
typedef int avec4 __attribute__((vector_size(16)));   // b128 transfer unit
__device__ __forceinline__ void async_copy_b128(const void* g, void* l) {
    __builtin_amdgcn_global_load_async_to_lds_b128((avec4*)g, (avec4*)l, 0, 0);
}
__device__ __forceinline__ void wait_async0() {
    __builtin_amdgcn_s_wait_asynccnt(0);
}
#endif

__device__ __forceinline__ unsigned pack_h2(float a, float b) {
    union { _Float16 h[2]; unsigned u; } u_;
    u_.h[0] = (_Float16)a;
    u_.h[1] = (_Float16)b;
    return u_.u;
}

// ---------------------------------------------------------------------------
// Kernel 1: per-row mean / inv-std of x.  One wave (32 lanes) per row.
// ---------------------------------------------------------------------------
__global__ __launch_bounds__(256) void stats_kernel(
    const float* __restrict__ x, float* __restrict__ mu, float* __restrict__ inv) {
    int row  = blockIdx.x * 8 + (threadIdx.x >> 5);
    int lane = threadIdx.x & 31;
    const float4* r4 = (const float4*)(x + (size_t)row * N_IN);
    float s = 0.f, ss = 0.f;
#pragma unroll
    for (int i = 0; i < 8; ++i) {              // 32 lanes * 8 * float4 = 1024
        float4 v = r4[lane + 32 * i];
        s  += v.x + v.y + v.z + v.w;
        ss += v.x * v.x + v.y * v.y + v.z * v.z + v.w * v.w;
    }
#pragma unroll
    for (int m = 16; m > 0; m >>= 1) {
        s  += __shfl_xor(s, m, 32);
        ss += __shfl_xor(ss, m, 32);
    }
    if (lane == 0) {
        float m_  = s * (1.0f / N_IN);
        float var = ss * (1.0f / N_IN) - m_ * m_;
        var = var < 0.f ? 0.f : var;
        mu[row]  = m_;
        inv[row] = 1.0f / (sqrtf(var) + EPS);
    }
}

// ---------------------------------------------------------------------------
// Kernel 2: circular 33-tap Gaussian conv of lm_raw rows -> f16 LMh, plus
// row sums S[p].  One block per p-row.
// ---------------------------------------------------------------------------
__global__ __launch_bounds__(256) void lm_build_kernel(
    const float* __restrict__ lm_raw, _Float16* __restrict__ LMh,
    float* __restrict__ S) {
    __shared__ float srow[N_IN];
    __shared__ float sfilt[FILT];
    __shared__ float sred[256];

    int p   = blockIdx.x;
    int tid = threadIdx.x;

    float4 v = ((const float4*)(lm_raw + (size_t)p * N_IN))[tid];
    ((float4*)srow)[tid] = v;
    sred[tid] = v.x + v.y + v.z + v.w;

    if (tid < FILT) {
        float ksum = 0.f, kt = 0.f;
        for (int t = 0; t < FILT; ++t) {
            float tt = (float)(t - FILT / 2) * (2.0f / (float)FILT) / SIGMA0;
            float k  = __expf(-0.5f * tt * tt);
            ksum += k;
            if (t == tid) kt = k;
        }
        sfilt[tid] = kt / ksum;
    }
    __syncthreads();

    for (int s = 128; s > 0; s >>= 1) {        // block reduce -> S[p]
        if (tid < s) sred[tid] += sred[tid + s];
        __syncthreads();
    }
    if (tid == 0) S[p] = sred[0];

#pragma unroll
    for (int j = 0; j < 4; ++j) {              // circular conv, 4 outputs/thread
        int n = tid * 4 + j;
        float acc = 0.f;
#pragma unroll
        for (int t = 0; t < FILT; ++t)
            acc += sfilt[t] * srow[(n - FILT / 2 + t) & (N_IN - 1)];
        LMh[(size_t)p * N_IN + n] = (_Float16)acc;
    }
}

// ---------------------------------------------------------------------------
// Kernel 3: double-buffered WMMA GEMM + fused normalization epilogue.
//   grid = (N/BN, M/BM), 256 threads = 8 waves (4 M x 2 N),
//   wave tile 32x32 via 2x2 v_wmma_f32_16x16x32_f16.
//   A: f32 global -> regs -> f16 LDS (conversion while staging).
//   B: f16 global -> LDS via global_load_async_to_lds_b128 (ASYNCcnt).
// ---------------------------------------------------------------------------
__global__ __launch_bounds__(256) void lift_gemm_kernel(
    const float* __restrict__ x, const _Float16* __restrict__ LMh,
    const float* __restrict__ mu, const float* __restrict__ inv,
    const float* __restrict__ S, float* __restrict__ out) {
    __shared__ __align__(16) _Float16 sA[2 * BM * SPAD];  // 2 x 128x32 (pad 40)
    __shared__ __align__(16) _Float16 sB[2 * BN * SPAD];  // 2 x  64x32 (pad 40)

    const int tid   = threadIdx.x;
    const int lane  = tid & 31;
    const int w     = tid >> 5;
    const int waveM = w >> 1;            // 0..3
    const int waveN = w & 1;             // 0..1
    const int half  = lane >> 4;         // 0/1
    const int l15   = lane & 15;
    const int blockN = blockIdx.x * BN;
    const int blockM = blockIdx.y * BM;

    // ---- A staging coords: fixed (row0, col4), rows stride 32 -------------
    const int ar0 = tid >> 3;            // 0..31
    const int ac4 = tid & 7;             // 0..7  (float4 column)
    const float* agp = x + (size_t)(blockM + ar0) * N_IN + ac4 * 4;
    unsigned* sAu = (unsigned*)sA;       // flattened; buf offset added at runtime
    const int aofs = ar0 * (SPAD / 2) + ac4 * 2;   // u32 index within a buffer

    // ---- B staging coords: one 16B chunk per thread -----------------------
    const int br = tid >> 2;             // 0..63
    const int bs = tid & 3;              // 0..3 (16B segment)
    const _Float16* bgp = LMh + (size_t)(blockN + br) * N_IN + bs * 8;
    const int bofs = br * SPAD + bs * 8; // half index within a buffer

    v8f acc[2][2] = {};
    float4 areg[4];

    auto gloadA = [&](int kb) {
#pragma unroll
        for (int i = 0; i < 4; ++i)
            areg[i] = *(const float4*)(agp + (size_t)i * 32 * N_IN + kb);
    };
    auto storeA = [&](int buf) {
        int o = buf * (BM * SPAD / 2) + aofs;
#pragma unroll
        for (int i = 0; i < 4; ++i) {
            sAu[o]     = pack_h2(areg[i].x, areg[i].y);
            sAu[o + 1] = pack_h2(areg[i].z, areg[i].w);
            o += 32 * (SPAD / 2);
        }
    };
    auto stageB = [&](int buf, int kb) {
        _Float16* dst = sB + buf * (BN * SPAD) + bofs;
#ifdef HAVE_ASYNC_LDS
        async_copy_b128(bgp + kb, dst);
#else
        const unsigned* g = (const unsigned*)(bgp + kb);
        unsigned* l = (unsigned*)dst;
        l[0] = g[0]; l[1] = g[1]; l[2] = g[2]; l[3] = g[3];
#endif
    };

    // Prologue: fill buffer 0
    stageB(0, 0);
    gloadA(0);
    storeA(0);
#ifdef HAVE_ASYNC_LDS
    wait_async0();
#endif
    __syncthreads();

    int cur = 0;
    for (int kb = 0; kb < N_IN; kb += BK) {
        const int  nxt      = cur ^ 1;
        const bool has_next = (kb + BK) < N_IN;
        if (has_next) {
            stageB(nxt, kb + BK);      // async B -> LDS[nxt]
            gloadA(kb + BK);           // A global loads in flight
        }

        // ---- fragments from LDS[cur] ----
        const _Float16* sAc = sA + cur * (BM * SPAD);
        const _Float16* sBc = sB + cur * (BN * SPAD);
        v16h afrag[2], bfrag[2];
#pragma unroll
        for (int m = 0; m < 2; ++m) {
            const _Float16* base = sAc + (waveM * 32 + m * 16 + l15) * SPAD;
            v8h a0 = *(const v8h*)(base + half * 8);
            v8h a1 = *(const v8h*)(base + 16 + half * 8);
            v16h a;
#pragma unroll
            for (int i = 0; i < 8; ++i) { a[i] = a0[i]; a[i + 8] = a1[i]; }
            afrag[m] = a;
        }
#pragma unroll
        for (int n = 0; n < 2; ++n) {
            const _Float16* base = sBc + (waveN * 32 + n * 16 + l15) * SPAD;
            v8h b0 = *(const v8h*)(base + half * 8);
            v8h b1 = *(const v8h*)(base + 16 + half * 8);
            v16h b;
#pragma unroll
            for (int i = 0; i < 8; ++i) { b[i] = b0[i]; b[i + 8] = b1[i]; }
            bfrag[n] = b;
        }

#pragma unroll
        for (int m = 0; m < 2; ++m)
#pragma unroll
            for (int n = 0; n < 2; ++n)
                acc[m][n] = __builtin_amdgcn_wmma_f32_16x16x32_f16(
                    false, afrag[m], false, bfrag[n],
                    (short)0, acc[m][n], false, false);

        if (has_next) storeA(nxt);     // f32->f16 convert + ds_store
#ifdef HAVE_ASYNC_LDS
        wait_async0();                 // B[nxt] resident before barrier
#endif
        __syncthreads();
        cur = nxt;
    }

    // Epilogue: y = inv_b * (acc - mu_b * S_p)
    const int b0 = blockM + waveM * 32;
    const int p0 = blockN + waveN * 32;
#pragma unroll
    for (int m = 0; m < 2; ++m) {
        int brow = b0 + m * 16 + half * 8;
#pragma unroll
        for (int n = 0; n < 2; ++n) {
            int p = p0 + n * 16 + l15;
            float Sp = S[p];
#pragma unroll
            for (int r = 0; r < 8; ++r) {
                int b = brow + r;
                out[(size_t)b * N_P + p] = inv[b] * (acc[m][n][r] - mu[b] * Sp);
            }
        }
    }
}

// ---------------------------------------------------------------------------
extern "C" void kernel_launch(void* const* d_in, const int* in_sizes, int n_in,
                              void* d_out, int out_size, void* d_ws, size_t ws_size,
                              hipStream_t stream) {
    const float* x      = (const float*)d_in[0];   // [16384, 1024] f32
    const float* lm_raw = (const float*)d_in[1];   // [32, 32, 1024] f32
    float*       out    = (float*)d_out;           // [16384, 32, 32] f32

    // Workspace layout (~2.23 MB)
    char* ws = (char*)d_ws;
    _Float16* LMh = (_Float16*)(ws);                        // 2 MiB
    float*    S   = (float*)(ws + 2 * 1024 * 1024);         // 4 KiB
    float*    mu  = (float*)(ws + 2 * 1024 * 1024 + 4096);  // 64 KiB
    float*    inv = (float*)(ws + 2 * 1024 * 1024 + 4096 + 65536);

    stats_kernel<<<M_ROWS / 8, 256, 0, stream>>>(x, mu, inv);
    lm_build_kernel<<<N_P, 256, 0, stream>>>(lm_raw, LMh, S);
    dim3 grid(N_P / BN, M_ROWS / BM);   // N-blocks adjacent -> A stripe L2-resident
    lift_gemm_kernel<<<grid, 256, 0, stream>>>(x, LMh, mu, inv, S, out);
}
